// RetinaNet_48713519072060
// MI455X (gfx1250) — matrix operand order repaired and checked by reading, not compiled, not run on hardware
//
#include <hip/hip_runtime.h>
#include <hip/hip_bf16.h>
#include <cstdint>

typedef __bf16 bf16_t;
typedef __attribute__((ext_vector_type(16))) __bf16 v16bf;
typedef __attribute__((ext_vector_type(8)))  __bf16 v8bf;
typedef __attribute__((ext_vector_type(8)))  float  v8f;
typedef int v4i __attribute__((vector_size(16)));   // matches builtin's param type

#define CIN        256
#define KTOT       2304      // 9 * 256
#define KSTEP      32
#define NK         (KTOT / KSTEP)   // 72
#define TILE_M     64
#define TILE_N     128
#define LDS_STRIDE 40        // bf16 elems; 80B rows keep 16B alignment, skew banks
#define ATILE      (TILE_M * LDS_STRIDE)   // 2560 bf16
#define BTILE      (TILE_N * LDS_STRIDE)   // 5120 bf16

#define AS1 __attribute__((address_space(1)))
#define AS3 __attribute__((address_space(3)))

#if __has_builtin(__builtin_amdgcn_global_load_async_to_lds_b128)
#define USE_ASYNC 1
#else
#define USE_ASYNC 0
#endif

__device__ __forceinline__ void wait_async0() {
#if __has_builtin(__builtin_amdgcn_s_wait_asynccnt)
    __builtin_amdgcn_s_wait_asynccnt(0);
#else
    asm volatile("s_wait_asynccnt 0x0" ::: "memory");
#endif
}

// ---------------------------------------------------------------------------
// Repack conv weights: f32 OIHW [O][256][3][3] -> bf16 [Opad][9][256]
// (k = tap*256 + c); rows >= O are zero-filled so GEMM loads are unconditional.
// ---------------------------------------------------------------------------
__global__ __launch_bounds__(256) void repack_w(const float* __restrict__ w,
                                                bf16_t* __restrict__ o,
                                                int O, int Opad) {
    long i = (long)blockIdx.x * 256 + threadIdx.x;
    if (i >= (long)Opad * KTOT) return;
    long oc = i / KTOT;
    int  k  = (int)(i - oc * KTOT);
    int  r  = k >> 8;
    int  c  = k & 255;
    float v = (oc < O) ? w[(oc * CIN + c) * 9 + r] : 0.0f;
    o[i] = (bf16_t)v;
}

// ---------------------------------------------------------------------------
// Convert input: f32 NCHW -> bf16 NHWC
// ---------------------------------------------------------------------------
__global__ __launch_bounds__(256) void nchw2nhwc(const float* __restrict__ x,
                                                 bf16_t* __restrict__ o,
                                                 int Nb, int H, int W) {
    long tot = (long)Nb * H * W * CIN;
    long i   = (long)blockIdx.x * 256 + threadIdx.x;
    if (i >= tot) return;
    int  c   = (int)(i & 255);
    long pix = i >> 8;
    long hw  = (long)H * W;
    long n   = pix / hw;
    long rem = pix - n * hw;
    o[i] = (bf16_t)x[(n * CIN + c) * hw + rem];
}

__global__ __launch_bounds__(256) void zero_bf16(bf16_t* __restrict__ p, int n) {
    int i = blockIdx.x * 256 + threadIdx.x;
    if (i < n) p[i] = (bf16_t)0.0f;
}

// ---------------------------------------------------------------------------
// 3x3 conv as implicit GEMM on WMMA bf16 (f32 accum).
//   Block tile: 64 (out chan) x 128 (positions); 8 waves; wave tile 32x32.
//   A = Wpad[o][k] bf16 (zero-padded rows) ; B = im2col[k][p] from NHWC bf16;
//   OOB B loads are redirected to a zero page -> no divergence in hot loop.
//   Double-buffered LDS fed by GLOBAL_LOAD_ASYNC_TO_LDS (ASYNCcnt) when
//   available; sync load/store fallback otherwise.
// ---------------------------------------------------------------------------
template <bool FINAL>
__global__ __launch_bounds__(256)
void conv3x3_bf16_wmma(const bf16_t* __restrict__ Xin,
                       const bf16_t* __restrict__ Wt,
                       const float*  __restrict__ bias,
                       const bf16_t* __restrict__ zp,
                       bf16_t* __restrict__ Yout,
                       float*  __restrict__ Fout,
                       int Nb, int H, int W, int O,
                       long aOff, long TA, int D) {
#if USE_ASYNC
    __shared__ __align__(16) bf16_t smem[(ATILE + BTILE) * 2];
#else
    __shared__ __align__(16) bf16_t smem[ATILE + BTILE];
#endif

    const int tid   = threadIdx.x;
    const int lane  = tid & 31;
    const int wave  = tid >> 5;
    const int waveM = wave & 1;        // 2 sub-tiles of 32 rows (channels)
    const int waveN = wave >> 1;       // 4 sub-tiles of 32 cols (positions)
    const int lrow  = lane & 15;
    const int half  = lane >> 4;

    const int HW    = H * W;
    const int P     = Nb * HW;
    const int pBase = blockIdx.x * TILE_N;
    const int oBase = blockIdx.y * TILE_M;

    // ---- loader mapping: A 64x32 (1 seg/thread), B 128x32 (2 segs/thread) ----
    const int arow = tid >> 2;            // 0..63 (A row, and B row group base)
    const int kseg = (tid & 3) * 8;       // 0,8,16,24

    const bf16_t* gA0 = Wt + (size_t)(oBase + arow) * KTOT + kseg;

    bool pv[2]; int py[2], px[2];
    const bf16_t* XnRow[2];
#pragma unroll
    for (int j = 0; j < 2; j++) {
        const int p = pBase + arow + j * 64;
        pv[j] = (p < P);
        int n = 0, yy = 0, xx = 0;
        if (pv[j]) {
            n = p / HW;
            const int pix = p - n * HW;
            yy = pix / W;
            xx = pix - yy * W;
        }
        py[j] = yy; px[j] = xx;
        XnRow[j] = Xin + (size_t)n * HW * CIN;
    }

    v8f acc00, acc01, acc10, acc11;
#pragma unroll
    for (int i = 0; i < 8; i++) { acc00[i] = 0.f; acc01[i] = 0.f; acc10[i] = 0.f; acc11[i] = 0.f; }

    // source pointers for one k-step (unconditional thanks to zero page)
    auto srcB = [&](int ks, int j) -> const bf16_t* {
        const int r  = ks >> 3;
        const int ry = (r * 86) >> 8;     // r/3 for r in [0,8]
        const int rx = r - ry * 3;
        const int cc = ((ks & 7) * 32) + kseg;
        const int iy = py[j] + ry - 1;
        const int ix = px[j] + rx - 1;
        const bool ok = pv[j] && (unsigned)iy < (unsigned)H && (unsigned)ix < (unsigned)W;
        return ok ? (XnRow[j] + (((size_t)iy * W + ix) << 8) + cc) : zp;
    };

    auto computeTile = [&](int buf) {
        const size_t sb = (size_t)buf * (ATILE + BTILE);
        v16bf a0, a1, b0, b1;
        {
            const bf16_t* ap = &smem[sb + (size_t)(waveM * 32 + lrow) * LDS_STRIDE + half * 8];
            v8bf lo = *(const v8bf*)ap, hi = *(const v8bf*)(ap + 16);
#pragma unroll
            for (int i = 0; i < 8; i++) { a0[i] = lo[i]; a0[8 + i] = hi[i]; }
            ap += 16 * LDS_STRIDE;
            lo = *(const v8bf*)ap; hi = *(const v8bf*)(ap + 16);
#pragma unroll
            for (int i = 0; i < 8; i++) { a1[i] = lo[i]; a1[8 + i] = hi[i]; }
        }
        {
            const bf16_t* bp = &smem[sb + ATILE + (size_t)(waveN * 32 + lrow) * LDS_STRIDE + half * 8];
            v8bf lo = *(const v8bf*)bp, hi = *(const v8bf*)(bp + 16);
#pragma unroll
            for (int i = 0; i < 8; i++) { b0[i] = lo[i]; b0[8 + i] = hi[i]; }
            bp += 16 * LDS_STRIDE;
            lo = *(const v8bf*)bp; hi = *(const v8bf*)(bp + 16);
#pragma unroll
            for (int i = 0; i < 8; i++) { b1[i] = lo[i]; b1[8 + i] = hi[i]; }
        }
        acc00 = __builtin_amdgcn_wmma_f32_16x16x32_bf16(false, a0, false, b0, (short)0, acc00, false, false);
        acc01 = __builtin_amdgcn_wmma_f32_16x16x32_bf16(false, a0, false, b1, (short)0, acc01, false, false);
        acc10 = __builtin_amdgcn_wmma_f32_16x16x32_bf16(false, a1, false, b0, (short)0, acc10, false, false);
        acc11 = __builtin_amdgcn_wmma_f32_16x16x32_bf16(false, a1, false, b1, (short)0, acc11, false, false);
    };

#if USE_ASYNC
    auto issueTile = [&](int ks, int buf) {
        const size_t sb = (size_t)buf * (ATILE + BTILE);
        __builtin_amdgcn_global_load_async_to_lds_b128(
            (AS1 v4i*)(gA0 + ks * 32),
            (AS3 v4i*)&smem[sb + (size_t)arow * LDS_STRIDE + kseg], 0, 0);
#pragma unroll
        for (int j = 0; j < 2; j++) {
            __builtin_amdgcn_global_load_async_to_lds_b128(
                (AS1 v4i*)srcB(ks, j),
                (AS3 v4i*)&smem[sb + ATILE + (size_t)(arow + j * 64) * LDS_STRIDE + kseg], 0, 0);
        }
    };

    issueTile(0, 0);
    for (int ks = 0; ks < NK; ++ks) {
        wait_async0();            // this wave's tile-ks DMA has landed in LDS
        __syncthreads();          // -> everyone's has; prev buffer free for reuse
        if (ks + 1 < NK) issueTile(ks + 1, (ks + 1) & 1);  // overlap with compute
        computeTile(ks & 1);
    }
#else
    for (int ks = 0; ks < NK; ++ks) {
        const v8bf av  = *(const v8bf*)(gA0 + ks * 32);
        const v8bf bv0 = *(const v8bf*)srcB(ks, 0);
        const v8bf bv1 = *(const v8bf*)srcB(ks, 1);
        __syncthreads();          // previous compute done before overwrite
        *(v8bf*)&smem[(size_t)arow * LDS_STRIDE + kseg]                 = av;
        *(v8bf*)&smem[ATILE + (size_t)arow * LDS_STRIDE + kseg]         = bv0;
        *(v8bf*)&smem[ATILE + (size_t)(arow + 64) * LDS_STRIDE + kseg]  = bv1;
        __syncthreads();
        computeTile(0);
    }
#endif

    // ---- epilogue: D tile (m,s); lane col n=lrow, VGPR v -> row 8*half+v ----
    const int ocm[2] = { oBase + waveM * 32 + half * 8,
                         oBase + waveM * 32 + 16 + half * 8 };
    float bs[2][8];
#pragma unroll
    for (int m = 0; m < 2; m++)
#pragma unroll
        for (int v = 0; v < 8; v++)
            bs[m][v] = (ocm[m] + v < O) ? bias[ocm[m] + v] : 0.0f;

#pragma unroll
    for (int m = 0; m < 2; m++) {
#pragma unroll
        for (int s = 0; s < 2; s++) {
            v8f acc = (m == 0) ? (s == 0 ? acc00 : acc01) : (s == 0 ? acc10 : acc11);
            const int p = pBase + waveN * 32 + s * 16 + lrow;
            if (p >= P) continue;
            const int oc0 = ocm[m];
            if (!FINAL) {
                v8bf ov;
#pragma unroll
                for (int v = 0; v < 8; v++) {
                    float f = acc[v] + bs[m][v];
                    ov[v] = (__bf16)fmaxf(f, 0.0f);
                }
                *(v8bf*)&Yout[(size_t)p * CIN + oc0] = ov;     // NHWC, 16B store
            } else {
                const int  n2   = p / HW;
                const int  pix  = p - n2 * HW;
                const long base = ((long)n2 * TA + aOff + (long)pix * 9) * D;
                if (oc0 + 7 < O) {
                    float4 f0 = make_float4(acc[0] + bs[m][0], acc[1] + bs[m][1],
                                            acc[2] + bs[m][2], acc[3] + bs[m][3]);
                    float4 f1 = make_float4(acc[4] + bs[m][4], acc[5] + bs[m][5],
                                            acc[6] + bs[m][6], acc[7] + bs[m][7]);
                    *(float4*)&Fout[base + oc0]     = f0;      // 16B aligned
                    *(float4*)&Fout[base + oc0 + 4] = f1;
                } else {
#pragma unroll
                    for (int v = 0; v < 8; v++) {
                        const int o = oc0 + v;
                        if (o < O) Fout[base + o] = acc[v] + bs[m][v];
                    }
                }
            }
        }
    }
}

// ---------------------------------------------------------------------------
// Host orchestration
// ---------------------------------------------------------------------------
extern "C" void kernel_launch(void* const* d_in, const int* in_sizes, int n_in,
                              void* d_out, int out_size, void* d_ws, size_t ws_size,
                              hipStream_t stream) {
    const float* feats[5] = {(const float*)d_in[0], (const float*)d_in[1],
                             (const float*)d_in[2], (const float*)d_in[3],
                             (const float*)d_in[4]};
    const float* clsW  = (const float*)d_in[5];
    const float* clsB  = (const float*)d_in[6];
    const float* clsOW = (const float*)d_in[7];
    const float* clsOB = (const float*)d_in[8];
    const float* regW  = (const float*)d_in[9];
    const float* regB  = (const float*)d_in[10];
    const float* regOW = (const float*)d_in[11];
    const float* regOB = (const float*)d_in[12];

    const int sizes[5] = {80, 40, 20, 10, 5};

    // workspace carve-up (bf16 elems); every offset is a multiple of 8 elems
    bf16_t* ws = (bf16_t*)d_ws;
    size_t off = 0;
    bf16_t* clsWb  = ws + off; off += (size_t)1024 * KTOT;   // 4 tower layers
    bf16_t* regWb  = ws + off; off += (size_t)1024 * KTOT;
    bf16_t* clsOWb = ws + off; off += (size_t)768 * KTOT;    // 720 -> pad 768
    bf16_t* regOWb = ws + off; off += (size_t)64 * KTOT;     // 36  -> pad 64
    bf16_t* zp     = ws + off; off += 128;                   // 256B zero page
    const size_t bufElems = (size_t)8 * 80 * 80 * 256;       // largest level
    bf16_t* bufIn = ws + off; off += bufElems;
    bf16_t* buf0  = ws + off; off += bufElems;
    bf16_t* buf1  = ws + off;

    zero_bf16<<<1, 256, 0, stream>>>(zp, 128);

    // repack all weights to bf16 [Opad][9][256] (zero-padded rows)
    {
        long t;
        t = 1024L * KTOT; repack_w<<<(t + 255) / 256, 256, 0, stream>>>(clsW,  clsWb,  1024, 1024);
        t = 1024L * KTOT; repack_w<<<(t + 255) / 256, 256, 0, stream>>>(regW,  regWb,  1024, 1024);
        t = 768L  * KTOT; repack_w<<<(t + 255) / 256, 256, 0, stream>>>(clsOW, clsOWb, 720, 768);
        t = 64L   * KTOT; repack_w<<<(t + 255) / 256, 256, 0, stream>>>(regOW, regOWb, 36, 64);
    }

    const long TA = 8525L * 9;                       // total anchors per image
    float* clsOut = (float*)d_out;
    float* regOut = clsOut + (size_t)8 * TA * 80;    // reg block follows cls

    long aOff = 0;
    for (int L = 0; L < 5; L++) {
        const int  H = sizes[L], W = sizes[L];
        const long P = 8L * H * W;

        {
            long tot = P * CIN;
            nchw2nhwc<<<(tot + 255) / 256, 256, 0, stream>>>(feats[L], bufIn, 8, H, W);
        }

        const dim3 blk(256);
        const unsigned gx = (unsigned)((P + TILE_N - 1) / TILE_N);
        const dim3 gTower(gx, 256 / TILE_M);   // y = 4
        const dim3 gCls(gx, 768 / TILE_M);     // y = 12
        const dim3 gReg(gx, 64 / TILE_M);      // y = 1

        // ---- cls tower + head ----
        {
            const bf16_t* src = bufIn;
            bf16_t* dst = buf0;
            for (int i = 0; i < 4; i++) {
                conv3x3_bf16_wmma<false><<<gTower, blk, 0, stream>>>(
                    src, clsWb + (size_t)i * 256 * KTOT, clsB + i * 256, zp,
                    dst, nullptr, 8, H, W, 256, 0, 0, 0);
                src = dst;
                dst = (dst == buf0) ? buf1 : buf0;
            }
            conv3x3_bf16_wmma<true><<<gCls, blk, 0, stream>>>(
                src, clsOWb, clsOB, zp, nullptr, clsOut, 8, H, W, 720, aOff, TA, 80);
        }

        // ---- reg tower + head ----
        {
            const bf16_t* src = bufIn;
            bf16_t* dst = buf0;
            for (int i = 0; i < 4; i++) {
                conv3x3_bf16_wmma<false><<<gTower, blk, 0, stream>>>(
                    src, regWb + (size_t)i * 256 * KTOT, regB + i * 256, zp,
                    dst, nullptr, 8, H, W, 256, 0, 0, 0);
                src = dst;
                dst = (dst == buf0) ? buf1 : buf0;
            }
            conv3x3_bf16_wmma<true><<<gReg, blk, 0, stream>>>(
                src, regOWb, regOB, zp, nullptr, regOut, 8, H, W, 36, aOff, TA, 4);
        }

        aOff += (long)H * W * 9;
    }
}
